// ScaledDotProductAttention_85847806313032
// MI455X (gfx1250) — matrix operand order repaired and checked by reading, not compiled, not run on hardware
//
#include <hip/hip_runtime.h>
#include <hip/hip_bf16.h>
#include <stdint.h>

#define Bv   8
#define Nv   2048
#define DM   512
#define DQK  64
#define DV   64

typedef __attribute__((ext_vector_type(16))) __bf16 v16bf;
typedef __attribute__((ext_vector_type(8)))  __bf16 v8bf;
typedef __attribute__((ext_vector_type(8)))  float  v8f;

__device__ __forceinline__ v16bf cat16(v8bf lo, v8bf hi) {
  v16bf r;
#pragma unroll
  for (int e = 0; e < 8; ++e) { r[e] = lo[e]; r[e + 8] = hi[e]; }
  return r;
}
// 16 contiguous bf16 (B operands: K row-major rows / Vt rows / W tiles in LDS)
__device__ __forceinline__ v16bf load16(const __bf16* p) {
  return cat16(*(const v8bf*)p, *(const v8bf*)(p + 8));
}

// ---------------------------------------------------------------------------
// Kernel 1: fused projections. Q,K row-major bf16 [B][N][64]; V transposed
// bf16 Vt [B][64][N]. A = X tile (16x512 in 32-chunks, f32->bf16), B = W tile
// staged per-k-chunk in LDS as [mat][n][k_local] so B-vectors are contiguous.
// ---------------------------------------------------------------------------
__global__ __launch_bounds__(128)
void proj_kernel(const float* __restrict__ x,
                 const float* __restrict__ wq,
                 const float* __restrict__ wk,
                 const float* __restrict__ wv,
                 __bf16* __restrict__ q,
                 __bf16* __restrict__ kmat,
                 __bf16* __restrict__ vt)
{
  __shared__ __bf16 wlds[3][64][32];   // 12 KB
  const int tid  = threadIdx.x;
  const int lane = tid & 31;
  const int wave = tid >> 5;           // 0..3 -> n-group
  const int half = lane >> 4;
  const int nloc = lane & 15;
  const int tile = blockIdx.x;         // 0..1023 row tiles of 16
  const int b    = tile >> 7;          // /128
  const int i0   = (tile & 127) << 4;

  // W staging pattern: n = tid&63 invariant, kl = (tid>>6) + 2u  (branch-free)
  const int n_ld = tid & 63;
  const int kl0  = tid >> 6;           // 0 or 1

  const float* xrow = x + ((size_t)b * Nv + i0 + nloc) * DM;

  v8f accq = {}, acck = {}, accv = {};

#pragma unroll 1
  for (int kk = 0; kk < DM / 32; ++kk) {
    __syncthreads();
#pragma unroll
    for (int m = 0; m < 3; ++m) {
      const float* wp = (m == 0) ? wq : (m == 1 ? wk : wv);  // constant per copy
#pragma unroll
      for (int u = 0; u < 16; ++u) {
        const int kl = kl0 + 2 * u;
        wlds[m][n_ld][kl] = (__bf16)wp[(size_t)(kk * 32 + kl) * 64 + n_ld];
      }
    }
    __syncthreads();

    // A operand: X rows, contiguous k order (k = 16*half + e)
    v8f xlo = *(const v8f*)(xrow + kk * 32 + half * 16);
    v8f xhi = *(const v8f*)(xrow + kk * 32 + half * 16 + 8);
    v16bf a;
#pragma unroll
    for (int e = 0; e < 8; ++e) { a[e] = (__bf16)xlo[e]; a[e + 8] = (__bf16)xhi[e]; }

    const __bf16* bq = &wlds[0][wave * 16 + nloc][half * 16];
    const __bf16* bk = &wlds[1][wave * 16 + nloc][half * 16];
    const __bf16* bv = &wlds[2][wave * 16 + nloc][half * 16];
    accq = __builtin_amdgcn_wmma_f32_16x16x32_bf16(false, a, false, load16(bq), (short)0, accq, false, false);
    acck = __builtin_amdgcn_wmma_f32_16x16x32_bf16(false, a, false, load16(bk), (short)0, acck, false, false);
    accv = __builtin_amdgcn_wmma_f32_16x16x32_bf16(false, a, false, load16(bv), (short)0, accv, false, false);
  }

  const int n = wave * 16 + nloc;
#pragma unroll
  for (int r = 0; r < 8; ++r) {
    const size_t row = (size_t)b * Nv + i0 + 8 * half + r;
    q[row * DQK + n]    = (__bf16)accq[r];
    kmat[row * DQK + n] = (__bf16)acck[r];
  }
  v8bf vv;
#pragma unroll
  for (int r = 0; r < 8; ++r) vv[r] = (__bf16)accv[r];
  *(v8bf*)&vt[((size_t)b * DV + n) * Nv + i0 + 8 * half] = vv;   // transposed store
}

// ---------------------------------------------------------------------------
// Kernel 2: flash attention. One wave per 16-row query tile; stream keys in
// 64-chunks (16 WMMA per iter) with online softmax; P: C/D->A via LDS bounce.
// ---------------------------------------------------------------------------
__global__ __launch_bounds__(128)
void attn_kernel(const unsigned char* __restrict__ mask,
                 const __bf16* __restrict__ q,
                 const __bf16* __restrict__ kmat,
                 const __bf16* __restrict__ vt,
                 float* __restrict__ out)
{
  __shared__ __bf16 pb[4][16][64];     // per-wave 16x64 P staging (8 KB)
  const int tid  = threadIdx.x;
  const int lane = tid & 31;
  const int wave = tid >> 5;
  const int half = lane >> 4;
  const int nloc = lane & 15;
  const int qt   = blockIdx.x * 4 + wave;   // 0..1023
  const int b    = qt >> 7;
  const int i0   = (qt & 127) << 4;
  const float C  = 0.125f * 1.44269504088896f;  // 1/sqrt(64) * log2(e)

  // Q tile in A layout, loaded once (two k-chunks)
  v16bf aq[2];
#pragma unroll
  for (int kk = 0; kk < 2; ++kk) {
    const __bf16* qp = q + ((size_t)b * Nv + i0 + nloc) * DQK + kk * 32;
    aq[kk] = cat16(*(const v8bf*)(qp + half * 16), *(const v8bf*)(qp + half * 16 + 8));
  }

  float mrow[8], lrow[8];
#pragma unroll
  for (int r = 0; r < 8; ++r) { mrow[r] = -3.0e38f; lrow[r] = 0.f; }
  v8f acc[4] = {{}, {}, {}, {}};

  const unsigned char* mb = mask + (size_t)b * Nv * Nv;

#pragma unroll 1
  for (int j0 = 0; j0 < Nv; j0 += 64) {
    // scores: four 16x16 tiles (j = j0 + t*16 + nloc), contraction over d=64
    v8f s[4] = {{}, {}, {}, {}};
#pragma unroll
    for (int t = 0; t < 4; ++t)
#pragma unroll
      for (int kk = 0; kk < 2; ++kk) {
        const __bf16* kp = kmat + ((size_t)b * Nv + j0 + t * 16 + nloc) * DQK + kk * 32 + half * 16;
        s[t] = __builtin_amdgcn_wmma_f32_16x16x32_bf16(false, aq[kk], false, load16(kp), (short)0, s[t], false, false);
      }

    // mask + chunk row-max (rows M=8*half+r live in element r across 16 lanes)
    bool  mk[4][8];
    float cmax[8];
#pragma unroll
    for (int r = 0; r < 8; ++r) {
      const unsigned char* mr = mb + (size_t)(i0 + 8 * half + r) * Nv + j0 + nloc;
      float mx = -3.0e38f;
#pragma unroll
      for (int t = 0; t < 4; ++t) {
        mk[t][r] = mr[t * 16] != 0;
        mx = fmaxf(mx, mk[t][r] ? s[t][r] : -3.0e38f);
      }
      cmax[r] = mx;
    }
#pragma unroll
    for (int r = 0; r < 8; ++r)
#pragma unroll
      for (int off = 1; off < 16; off <<= 1)
        cmax[r] = fmaxf(cmax[r], __shfl_xor(cmax[r], off, 32));

    float alpha[8], rsum[8];
#pragma unroll
    for (int r = 0; r < 8; ++r) {
      const float mn = fmaxf(mrow[r], cmax[r]);
      alpha[r] = __builtin_amdgcn_exp2f((mrow[r] - mn) * C);
      float rs = 0.f;
#pragma unroll
      for (int t = 0; t < 4; ++t) {
        const float p = mk[t][r] ? __builtin_amdgcn_exp2f((s[t][r] - mn) * C) : 0.f;
        s[t][r] = p;
        rs += p;
      }
      rsum[r] = rs;
      mrow[r] = mn;
    }
#pragma unroll
    for (int r = 0; r < 8; ++r) {
#pragma unroll
      for (int off = 1; off < 16; off <<= 1)
        rsum[r] += __shfl_xor(rsum[r], off, 32);
      lrow[r] = lrow[r] * alpha[r] + rsum[r];
    }
#pragma unroll
    for (int g = 0; g < 4; ++g)
#pragma unroll
      for (int r = 0; r < 8; ++r) acc[g][r] *= alpha[r];

    // P: C/D layout -> LDS (row-major 16x64 bf16) -> A layout (two 16x32 ops)
#pragma unroll
    for (int t = 0; t < 4; ++t)
#pragma unroll
      for (int r = 0; r < 8; ++r)
        pb[wave][8 * half + r][t * 16 + nloc] = (__bf16)s[t][r];
    asm volatile("" ::: "memory");   // keep DS store->load order (same-wave DS is in-order)
    v16bf pa[2];
#pragma unroll
    for (int kk = 0; kk < 2; ++kk) {
      const __bf16* pr = &pb[wave][nloc][kk * 32 + half * 16];
      pa[kk] = cat16(*(const v8bf*)pr, *(const v8bf*)(pr + 8));
    }

    // O += P @ V : B operand from transposed V -> contiguous per lane
#pragma unroll
    for (int g = 0; g < 4; ++g)
#pragma unroll
      for (int kk = 0; kk < 2; ++kk) {
        const __bf16* vp = vt + ((size_t)b * DV + g * 16 + nloc) * Nv + j0 + kk * 32 + half * 16;
        acc[g] = __builtin_amdgcn_wmma_f32_16x16x32_bf16(false, pa[kk], false, load16(vp), (short)0, acc[g], false, false);
      }

    if (j0 + 64 < Nv) {
      __builtin_prefetch(kmat + ((size_t)b * Nv + j0 + 64 + lane) * DQK, 0, 1);
      __builtin_prefetch(vt + ((size_t)b * DV + lane) * Nv + j0 + 64, 0, 1);
    }
  }

#pragma unroll
  for (int r = 0; r < 8; ++r) {
    const float inv = 1.0f / fmaxf(lrow[r], 1e-30f);
    const size_t row = (size_t)b * Nv + i0 + 8 * half + r;
#pragma unroll
    for (int g = 0; g < 4; ++g)
      out[row * DV + g * 16 + nloc] = acc[g][r] * inv;
  }
}

// ---------------------------------------------------------------------------
extern "C" void kernel_launch(void* const* d_in, const int* in_sizes, int n_in,
                              void* d_out, int out_size, void* d_ws, size_t ws_size,
                              hipStream_t stream) {
  (void)in_sizes; (void)n_in; (void)out_size; (void)ws_size;
  const unsigned char* mask = (const unsigned char*)d_in[0];  // bool [B,N,N]
  const float* x  = (const float*)d_in[1];                    // [B,N,512]
  const float* wk = (const float*)d_in[2];                    // [512,64]
  const float* wq = (const float*)d_in[3];                    // [512,64]
  const float* wv = (const float*)d_in[4];                    // [512,64]
  float* out = (float*)d_out;                                 // [B,N,64] f32

  __bf16* qb  = (__bf16*)d_ws;                                // 2 MB
  __bf16* kb  = qb + (size_t)Bv * Nv * DQK;                   // 2 MB
  __bf16* vtb = kb + (size_t)Bv * Nv * DQK;                   // 2 MB (transposed)

  proj_kernel<<<dim3(Bv * Nv / 16), dim3(128), 0, stream>>>(x, wq, wk, wv, qb, kb, vtb);
  attn_kernel<<<dim3(Bv * Nv / 16 / 4), dim3(128), 0, stream>>>(mask, qb, kb, vtb, out);
}
